// Net_83141976916002
// MI455X (gfx1250) — compile-verified, matrix operand order checked by reading, not run on hardware
//
#include <hip/hip_runtime.h>
#include <math.h>

typedef __attribute__((ext_vector_type(16))) _Float16 v16h;
typedef __attribute__((ext_vector_type(4)))  _Float16 v4h;
typedef __attribute__((ext_vector_type(8)))  float    v8f;

#define LN_EPS 1e-5f
#define NB   32      // batch
#define NL   6       // layers
#define NE   3       // embed
#define NF   2048    // mlp hidden
#define NHW  32

// Branchless exact GELU: erf via Abramowitz-Stegun 7.1.26 (|err|<=1.5e-7).
// One v_rcp + one v_exp + ~12 VALU, no divergence.
__device__ __forceinline__ float gelu_exact(float x) {
  const float z = fabsf(x) * 0.70710678118654752f;       // |x|/sqrt(2)
  const float t = __builtin_amdgcn_rcpf(fmaf(0.3275911f, z, 1.0f));
  float poly = fmaf(1.061405429f, t, -1.453152027f);
  poly = fmaf(poly, t, 1.421413741f);
  poly = fmaf(poly, t, -0.284496736f);
  poly = fmaf(poly, t, 0.254829592f);
  poly *= t;
  const float e = __expf(-z * z);
  const float erf_abs = fmaf(-poly, e, 1.0f);
  const float erf_s = copysignf(erf_abs, x);
  return 0.5f * x * (1.0f + erf_s);
}

// ---------------------------------------------------------------------------
// Kernel 0: pack MLP weights once per launch.
//  pw1[l*NF+col]  = f16x4 {W1[l,col,0..2], 0}   -> B-operand built by one b64 load
//  pbw2[l*NF+col] = f32x4 {b1, W2[e=0], W2[e=1], W2[e=2]} -> one b128 load
// ---------------------------------------------------------------------------
__global__ void __launch_bounds__(256) k_pack(const float* __restrict__ W1,
                                              const float* __restrict__ b1,
                                              const float* __restrict__ W2,
                                              v4h* __restrict__ pw1,
                                              float4* __restrict__ pbw2) {
  const int idx = blockIdx.x * 256 + threadIdx.x;   // l*NF + col
  if (idx < NL * NF) {
    const int l = idx >> 11, col = idx & (NF - 1);
    const float* w1r = W1 + idx * 3;
    v4h w;
    w[0] = (_Float16)w1r[0];
    w[1] = (_Float16)w1r[1];
    w[2] = (_Float16)w1r[2];
    w[3] = (_Float16)0.f;
    pw1[idx] = w;
    float4 c;
    c.x = b1[idx];
    c.y = W2[(l * 3 + 0) * NF + col];
    c.z = W2[(l * 3 + 1) * NF + col];
    c.w = W2[(l * 3 + 2) * NF + col];
    pbw2[idx] = c;
  }
}

// ---------------------------------------------------------------------------
// Kernel 1: per-(b,ch) conv dot: c[b,ch] = sum(x[b,ch]*conv_w[ch]) + conv_b[ch]
// ---------------------------------------------------------------------------
__global__ void __launch_bounds__(128) k_conv(const float* __restrict__ x,
                                              const float* __restrict__ cw,
                                              const float* __restrict__ cb,
                                              float* __restrict__ c_ws) {
  __shared__ float red[3][128];
  const int b = blockIdx.x, tid = threadIdx.x;
  #pragma unroll
  for (int ch = 0; ch < 3; ++ch) {
    float p = 0.f;
    for (int i = tid; i < 1024; i += 128)
      p += x[b * 3072 + ch * 1024 + i] * cw[ch * 1024 + i];
    red[ch][tid] = p;
  }
  __syncthreads();
  if (tid < 3) {
    float s = 0.f;
    for (int i = 0; i < 128; ++i) s += red[tid][i];   // fixed order -> deterministic
    c_ws[b * 3 + tid] = s + cb[tid];
  }
}

// ---------------------------------------------------------------------------
// Kernel 2: one block per (b, n) slice: 32 tokens through all 6 layers.
//   wave 0 -> tokens 0..15, wave 1 -> tokens 16..31 for the WMMA MLP.
// ---------------------------------------------------------------------------
__global__ void __launch_bounds__(64)
k_main(const float* __restrict__ x,     const float* __restrict__ c_ws,
       const float* __restrict__ Wqkv,  const float* __restrict__ bqkv,
       const float* __restrict__ Wo,    const float* __restrict__ bo,
       const v4h*   __restrict__ pw1,   const float4* __restrict__ pbw2,
       const float* __restrict__ b2,
       const float* __restrict__ ln1g,  const float* __restrict__ ln1b,
       const float* __restrict__ ln2g,  const float* __restrict__ ln2b,
       const float* __restrict__ mlpW,  float* __restrict__ part_ws) {
  __shared__ float sh[32][3];    // h for the 32 local tokens
  __shared__ float sqkv[32][9];  // q(0..2) k(3..5) v(6..8) per token

  const int blk  = blockIdx.x;
  const int b    = blk >> 5;
  const int n    = blk & 31;
  const int tid  = threadIdx.x;
  const int lane = tid & 31;
  const int wave = tid >> 5;

  // ---- build h:  h[s][e] = x[b, i] + c[b, i>>10],  i = 96*s + 3*n + e ----
  if (tid < 32) {
    const int s = tid;
    #pragma unroll
    for (int e = 0; e < 3; ++e) {
      const int i = 96 * s + 3 * n + e;
      sh[s][e] = x[b * 3072 + i] + c_ws[b * 3 + (i >> 10)];
    }
  }
  __syncthreads();

  for (int l = 0; l < NL; ++l) {
    // ---- QKV: 3x3 matvecs per token ----
    if (tid < 32) {
      const int s = tid;
      const float h0 = sh[s][0], h1 = sh[s][1], h2 = sh[s][2];
      const float* W  = Wqkv + l * 27;
      const float* bb = bqkv + l * 9;
      #pragma unroll
      for (int f = 0; f < 9; ++f)
        sqkv[s][f] = W[f * 3 + 0] * h0 + W[f * 3 + 1] * h1 + W[f * 3 + 2] * h2 + bb[f];
    }
    __syncthreads();

    // ---- attention (head_dim = 1): item = (head hh, row s); softmax over t ----
    for (int idx = tid; idx < 96; idx += 64) {
      const int hh = idx >> 5, s = idx & 31;
      const float q = sqkv[s][hh];
      float m = -INFINITY, sum = 0.f, acc = 0.f;
      for (int t = 0; t < 32; ++t) {
        const float sc = q * sqkv[t][3 + hh];
        const float mn = fmaxf(m, sc);
        const float e0 = __expf(m - mn);
        const float e1 = __expf(sc - mn);
        sum = sum * e0 + e1;
        acc = acc * e0 + e1 * sqkv[t][6 + hh];
        m = mn;
      }
      sqkv[s][hh] = acc / sum;   // overwrite own q slot (only this item touches it)
    }
    __syncthreads();

    // ---- Wo matvec + residual + LN1 ----
    if (tid < 32) {
      const int s = tid;
      const float o0 = sqkv[s][0], o1 = sqkv[s][1], o2 = sqkv[s][2];
      const float* W  = Wo + l * 9;
      const float* bb = bo + l * 3;
      float r[3];
      #pragma unroll
      for (int f = 0; f < 3; ++f)
        r[f] = sh[s][f] + W[f * 3 + 0] * o0 + W[f * 3 + 1] * o1 + W[f * 3 + 2] * o2 + bb[f];
      const float mu = (r[0] + r[1] + r[2]) * (1.f / 3.f);
      const float d0 = r[0] - mu, d1 = r[1] - mu, d2 = r[2] - mu;
      const float var = (d0 * d0 + d1 * d1 + d2 * d2) * (1.f / 3.f);
      const float inv = 1.f / sqrtf(var + LN_EPS);
      sh[s][0] = d0 * inv * ln1g[l * 3 + 0] + ln1b[l * 3 + 0];
      sh[s][1] = d1 * inv * ln1g[l * 3 + 1] + ln1b[l * 3 + 1];
      sh[s][2] = d2 * inv * ln1g[l * 3 + 2] + ln1b[l * 3 + 2];
    }
    __syncthreads();

    // ---- MLP via WMMA. GEMM1: [16 tok x K=3(pad32)] x [K x 16 of F] per tile.
    //      GEMM2 (N=3) fused: per-lane FMA against W2 column + shfl reduce. ----
    {
      const int s0 = wave * 16;
      // A operand: lanes 0-15 carry M=lane, K=0..2 in elements 0..2 (rest 0).
      v16h A = {};
      if (lane < 16) {
        A[0] = (_Float16)sh[s0 + lane][0];
        A[1] = (_Float16)sh[s0 + lane][1];
        A[2] = (_Float16)sh[s0 + lane][2];
      }
      float p[8][3];
      #pragma unroll
      for (int v = 0; v < 8; ++v) { p[v][0] = 0.f; p[v][1] = 0.f; p[v][2] = 0.f; }

      const int nl = lane & 15;                  // this lane's output column in the tile
      const v4h*    pw1l  = pw1  + l * NF;
      const float4* pbw2l = pbw2 + l * NF;

      #pragma unroll 2
      for (int n0 = 0; n0 < NF; n0 += 16) {
        const int ncol = n0 + nl;
        // B operand: lanes 0-15 carry K=0..15 across elements; only K=0..2 nonzero.
        v16h Bm = {};
        if (lane < 16) {
          const v4h w = pw1l[ncol];
          Bm[0] = w[0];
          Bm[1] = w[1];
          Bm[2] = w[2];
        }
        v8f C = {};
        C = __builtin_amdgcn_wmma_f32_16x16x32_f16(false, A, false, Bm,
                                                   (short)0, C, false, false);
        // C layout: lane -> column (lane&15), VGPR v -> row v + 8*(lane>=16)
        const float4 cw = pbw2l[ncol];   // {b1, W2[0], W2[1], W2[2]}
        #pragma unroll
        for (int v = 0; v < 8; ++v) {
          const float g = gelu_exact(C[v] + cw.x);
          p[v][0] += g * cw.y;
          p[v][1] += g * cw.z;
          p[v][2] += g * cw.w;
        }
      }

      // reduce partials across the 16 lanes of each half-wave (tile columns)
      #pragma unroll
      for (int v = 0; v < 8; ++v)
        #pragma unroll
        for (int e = 0; e < 3; ++e) {
          float val = p[v][e];
          val += __shfl_xor(val, 8, 32);
          val += __shfl_xor(val, 4, 32);
          val += __shfl_xor(val, 2, 32);
          val += __shfl_xor(val, 1, 32);
          p[v][e] = val;
        }

      // lane 0 -> rows 0..7, lane 16 -> rows 8..15: residual + b2 + LN2
      if ((lane & 15) == 0) {
        const int mbase = (lane >> 4) * 8;
        for (int v = 0; v < 8; ++v) {
          const int s = s0 + mbase + v;
          float r[3];
          #pragma unroll
          for (int e = 0; e < 3; ++e)
            r[e] = sh[s][e] + p[v][e] + b2[l * 3 + e];
          const float mu = (r[0] + r[1] + r[2]) * (1.f / 3.f);
          const float d0 = r[0] - mu, d1 = r[1] - mu, d2 = r[2] - mu;
          const float var = (d0 * d0 + d1 * d1 + d2 * d2) * (1.f / 3.f);
          const float inv = 1.f / sqrtf(var + LN_EPS);
          sh[s][0] = d0 * inv * ln2g[l * 3 + 0] + ln2b[l * 3 + 0];
          sh[s][1] = d1 * inv * ln2g[l * 3 + 1] + ln2b[l * 3 + 1];
          sh[s][2] = d2 * inv * ln2g[l * 3 + 2] + ln2b[l * 3 + 2];
        }
      }
    }
    __syncthreads();
  }

  // ---- head partials: part[b,n,j] = sum_{s,e} h[s][e] * mlpW[j, 96s+3n+e] ----
  if (tid < 32) {
    const int s = tid;
    float acc[10];
    #pragma unroll
    for (int j = 0; j < 10; ++j) acc[j] = 0.f;
    #pragma unroll
    for (int e = 0; e < 3; ++e) {
      const float hv = sh[s][e];
      const int   i  = 96 * s + 3 * n + e;
      #pragma unroll
      for (int j = 0; j < 10; ++j) acc[j] += hv * mlpW[j * 3072 + i];
    }
    #pragma unroll
    for (int j = 0; j < 10; ++j) {
      float val = acc[j];
      val += __shfl_xor(val, 16, 32);
      val += __shfl_xor(val, 8, 32);
      val += __shfl_xor(val, 4, 32);
      val += __shfl_xor(val, 2, 32);
      val += __shfl_xor(val, 1, 32);
      if (lane == 0) part_ws[blk * 10 + j] = val;
    }
  }
}

// ---------------------------------------------------------------------------
// Kernel 3: deterministic final reduction over n: out[b,j] = mlp_b[j] + sum_n part
// ---------------------------------------------------------------------------
__global__ void __launch_bounds__(320) k_head(const float* __restrict__ part_ws,
                                              const float* __restrict__ mlpb,
                                              float* __restrict__ out) {
  const int t = threadIdx.x;
  if (t < 320) {
    const int b = t / 10, j = t % 10;
    float s = mlpb[j];
    for (int nn = 0; nn < 32; ++nn)       // fixed order -> deterministic
      s += part_ws[(b * 32 + nn) * 10 + j];
    out[t] = s;
  }
}

extern "C" void kernel_launch(void* const* d_in, const int* in_sizes, int n_in,
                              void* d_out, int out_size, void* d_ws, size_t ws_size,
                              hipStream_t stream) {
  (void)in_sizes; (void)n_in; (void)out_size; (void)ws_size;
  const float* x     = (const float*)d_in[0];
  const float* convw = (const float*)d_in[1];
  const float* convb = (const float*)d_in[2];
  const float* Wqkv  = (const float*)d_in[3];
  const float* bqkv  = (const float*)d_in[4];
  const float* Wo    = (const float*)d_in[5];
  const float* bo    = (const float*)d_in[6];
  const float* W1    = (const float*)d_in[7];
  const float* b1    = (const float*)d_in[8];
  const float* W2    = (const float*)d_in[9];
  const float* b2    = (const float*)d_in[10];
  const float* ln1g  = (const float*)d_in[11];
  const float* ln1b  = (const float*)d_in[12];
  const float* ln2g  = (const float*)d_in[13];
  const float* ln2b  = (const float*)d_in[14];
  const float* mlpW  = (const float*)d_in[15];
  const float* mlpb  = (const float*)d_in[16];

  char* ws = (char*)d_ws;
  float*  c_ws    = (float*)(ws);                 //   1 KB region (96 used)
  float*  part_ws = (float*)(ws + 1024);          //  40 KB (10240 floats)
  v4h*    pw1     = (v4h*)(ws + 48 * 1024);       //  96 KB (12288 * 8 B)
  float4* pbw2    = (float4*)(ws + 160 * 1024);   // 192 KB (12288 * 16 B)

  k_pack<<<(NL * NF + 255) / 256, 256, 0, stream>>>(W1, b1, W2, pw1, pbw2);
  k_conv<<<NB, 128, 0, stream>>>(x, convw, convb, c_ws);
  k_main<<<NB * NHW, 64, 0, stream>>>(x, c_ws, Wqkv, bqkv, Wo, bo, pw1, pbw2, b2,
                                      ln1g, ln1b, ln2g, ln2b, mlpW, part_ws);
  k_head<<<1, 320, 0, stream>>>(part_ws, mlpb, (float*)d_out);
}
